// PAM_Module_51007031607712
// MI455X (gfx1250) — matrix operand order repaired and checked by reading, not compiled, not run on hardware
//
#include <hip/hip_runtime.h>
#include <hip/hip_bf16.h>

// ---------------------------------------------------------------------------
// PAM module for MI455X (gfx1250), wave32 + WMMA bf16 (f32 accumulate).
//   B=8, C=512, H=W=96 -> N=9216, CQ=64
// Refactored: matrix = (K_hat X^T) Wv^T ; v_sum = Wv colsum(x) + N*bv
// eliminating the 39-GFLOP V GEMM and its 151 MB intermediate.
// qk_kernel: double-buffered async LDS transpose staging
// (global_load_async_to_lds_b32 + s_wait_asynccnt 16) with a single
// cooperative f32->bf16 LDS convert pass (double-buffered output tile), so
// WMMA operands come straight from ds_load_b128 with no VALU in the chain.
// tgemm_kernel: folds the x column-sum (for v_sum) into its f32 B loads.
// ---------------------------------------------------------------------------

#define BATCH 8
#define CCH   512
#define CQ    64
#define NPOS  9216
#define EPSV  1e-6f
#define XLD   68      // padded f32 LDS row stride (floats)
#define XTB   72      // padded bf16 LDS row stride (bf16 elems), 144B = 9*16B

typedef __bf16 bf16_t;
typedef bf16_t v16bf __attribute__((ext_vector_type(16)));
typedef bf16_t v8bf  __attribute__((ext_vector_type(8)));
typedef float  v8f   __attribute__((ext_vector_type(8)));
typedef float  v4f   __attribute__((ext_vector_type(4)));

union frag16 { v16bf v; v8bf h[2]; };

// ---- WMMA wrapper: D = A(16x32 bf16) * B(32x16 bf16) + C(16x16 f32) --------
__device__ __forceinline__ v8f wmma_bf16(v16bf a, v16bf b, v8f c) {
  return __builtin_amdgcn_wmma_f32_16x16x32_bf16(
      /*neg_a=*/false, a, /*neg_b=*/false, b,
      /*c_mod=*/(short)0, c, /*reuse_a=*/false, /*reuse_b=*/false);
}

// A-matrix 16x32 bf16 per-lane layout used here (contiguous k, not the ISA table order; B matches):
//   lane L: row m = L&15 ; K = kb..kb+15, kb=(L>>4)*16
__device__ __forceinline__ v16bf load_frag_a(const bf16_t* p, int ld, int lane) {
  int m  = lane & 15;
  int kb = (lane >> 4) * 16;
  frag16 u;
  u.h[0] = *(const v8bf*)(p + (size_t)m * ld + kb);
  u.h[1] = *(const v8bf*)(p + (size_t)m * ld + kb + 8);
  return u.v;
}

// B-matrix 32x16 bf16: lane L: col n = L&15 ; K = kb..kb+15, kb=(L>>4)*16.
// Source stored K-contiguous per column ([n][k], stride ld).
__device__ __forceinline__ v16bf load_frag_b(const bf16_t* p, int ld, int lane) {
  int n  = lane & 15;
  int kb = (lane >> 4) * 16;
  frag16 u;
  u.h[0] = *(const v8bf*)(p + (size_t)n * ld + kb);
  u.h[1] = *(const v8bf*)(p + (size_t)n * ld + kb + 8);
  return u.v;
}

// ---------------------------------------------------------------------------
__global__ void zero_kernel(float* p, int count) {
  int i = blockIdx.x * 256 + threadIdx.x;
  if (i < count) p[i] = 0.0f;
}

__global__ void wcvt_kernel(const float* wq, const float* wk,
                            bf16_t* wqb, bf16_t* wkb) {
  int i = blockIdx.x * 256 + threadIdx.x;  // 64*512 = 32768 elems
  if (i < CQ * CCH) {
    wqb[i] = (bf16_t)wq[i];
    wkb[i] = (bf16_t)wk[i];
  }
}

// ---------------------------------------------------------------------------
// Q/K projection + L2 normalize over channel dim.
//   grid = B * (NPOS/64) blocks, 256 threads (8 waves).
//   waves 0-3: Q m-tiles 0..3 ; waves 4-7: K m-tiles 0..3.
// Per 64-c K-step: async transpose-stage f32 (double buffered, prefetch one
// step ahead), one cooperative cvt pass to bf16 LDS (also double buffered),
// then pure-LDS bf16 fragments -> WMMA.
// Outputs: Qbt[b][n][64] bf16 (transposed), Kb[b][64][N] bf16,
//          ksum[b][m] += sum_n Khat (f32 atomics)
// ---------------------------------------------------------------------------
__device__ __forceinline__ void stage_async64(const float* xb, int c0, int n0,
                                              unsigned ldsb, int t) {
#pragma unroll
  for (int i = 0; i < 16; ++i) {
    int idx = t + i * 256;              // 0..4095 : LDS[nn][cc] <- x[c0+cc][n0+nn]
    int nn = idx & 63, cc = idx >> 6;
    const float* gp = xb + (size_t)(c0 + cc) * NPOS + n0 + nn;
    unsigned la = ldsb + (unsigned)((nn * XLD + cc) * 4);
    asm volatile("global_load_async_to_lds_b32 %0, %1, off"
                 :: "v"(la), "v"((unsigned long long)(uintptr_t)gp)
                 : "memory");
  }
}

__global__ void qk_kernel(const float* __restrict__ x,
                          const bf16_t* __restrict__ wqb,
                          const bf16_t* __restrict__ wkb,
                          const float* __restrict__ bq,
                          const float* __restrict__ bk,
                          bf16_t* Qbt, bf16_t* Kb, float* ksum) {
  int b  = blockIdx.x / (NPOS / 64);
  int n0 = (blockIdx.x % (NPOS / 64)) * 64;
  const float* xb = x + (size_t)b * CCH * NPOS;
  int t = threadIdx.x, wave = t >> 5, lane = t & 31;

  __shared__ __align__(16) float  xf[2 * 64 * XLD];    // f32 double buffer
  __shared__ __align__(16) bf16_t xtb[2 * 64 * XTB];   // bf16 double buffer
  __shared__ float qk[2][64 * 64];                     // raw Q ([0]) / K ([1])
  __shared__ float rnorm[2][64];

  bool isK  = wave >= 4;
  int mtile = wave & 3;
  const bf16_t* W = isK ? wkb : wqb;
  unsigned base0 = (unsigned)(uintptr_t)&xf[0];
  unsigned base1 = (unsigned)(uintptr_t)&xf[64 * XLD];

  stage_async64(xb, 0, n0, base0, t);                // prefetch tile 0

  v8f acc[4] = {};
  for (int it = 0; it < 8; ++it) {
    int c0 = it * 64;
    if (it < 7) {                                    // prefetch next tile
      stage_async64(xb, c0 + 64, n0, (it & 1) ? base0 : base1, t);
      asm volatile("s_wait_asynccnt 0x10" ::: "memory");  // older 16 done
    } else {
      asm volatile("s_wait_asynccnt 0x0" ::: "memory");
    }
    __syncthreads();                                 // tile `it` in LDS
    // ---- cooperative convert pass: xf[it&1] -> xtb[it&1] ----
    bf16_t* tile = xtb + (it & 1) * (64 * XTB);
    {
      const float* cur = xf + (it & 1) * (64 * XLD);
      int nn = t & 63, cb = (t >> 6) * 16;
      const float* s4 = cur + nn * XLD + cb;
      v4f f0 = *(const v4f*)(s4);
      v4f f1 = *(const v4f*)(s4 + 4);
      v4f f2 = *(const v4f*)(s4 + 8);
      v4f f3 = *(const v4f*)(s4 + 12);
      __align__(16) bf16_t tb[16];
#pragma unroll
      for (int i = 0; i < 4; ++i) {
        tb[i]      = (bf16_t)f0[i];
        tb[4 + i]  = (bf16_t)f1[i];
        tb[8 + i]  = (bf16_t)f2[i];
        tb[12 + i] = (bf16_t)f3[i];
      }
      bf16_t* d = tile + nn * XTB + cb;
      *(v8bf*)(d)     = *(const v8bf*)(tb);
      *(v8bf*)(d + 8) = *(const v8bf*)(tb + 8);      // 16B offset
    }
    __syncthreads();
    // ---- 2 K-substeps x 4 n-subtiles, A fragment reused across n ----
    // (no trailing barrier: convert(it+2) reuses this tile buffer and is
    //  separated from these reads by two barriers)
#pragma unroll
    for (int s = 0; s < 2; ++s) {
      v16bf a = load_frag_a(W + (size_t)(mtile * 16) * CCH + c0 + s * 32,
                            CCH, lane);
#pragma unroll
      for (int nt = 0; nt < 4; ++nt) {
        v16bf bfr = load_frag_b(tile + (nt * 16) * XTB + s * 32, XTB, lane);
        acc[nt] = wmma_bf16(a, bfr, acc[nt]);
      }
    }
  }
  __syncthreads();

  // C/D layout: VGPR r -> M = (lane>=16 ? 8 : 0)+r, N = lane&15
  {
    int nlo = lane & 15, mhi = (lane >> 4) * 8;
#pragma unroll
    for (int nt = 0; nt < 4; ++nt)
#pragma unroll
      for (int r = 0; r < 8; ++r) {
        int m = mtile * 16 + mhi + r;
        float bias = isK ? bk[m] : bq[m];
        qk[isK ? 1 : 0][m * 64 + nt * 16 + nlo] = acc[nt][r] + bias;
      }
  }
  __syncthreads();

  if (t < 128) {                                 // column L2 norms
    int which = t >> 6, n = t & 63;
    float s = 0.0f;
    for (int m = 0; m < 64; ++m) {
      float v = qk[which][m * 64 + n];
      s += v * v;
    }
    rnorm[which][n] = rsqrtf(s);
  }
  __syncthreads();

  // Q_hat -> Qbt[b][n][m] (n-major, 64 contiguous m): coalesced 32B stores
  {
    int n = t >> 2, mb = (t & 3) * 16;
    float rq = rnorm[0][n];
    __align__(16) bf16_t tmp[16];
#pragma unroll
    for (int i = 0; i < 16; ++i)
      tmp[i] = (bf16_t)(qk[0][(mb + i) * 64 + n] * rq);
    bf16_t* dst = Qbt + ((size_t)b * NPOS + n0 + n) * 64 + mb;
    *(v8bf*)(dst)     = *(const v8bf*)(tmp);
    *(v8bf*)(dst + 8) = *(const v8bf*)(tmp + 8);
  }
  // K_hat -> Kb[b][m][N] (m-major, contiguous n) + ksum atomics
  {
    int m = t >> 2, nb = (t & 3) * 16;
    float part = 0.0f;
    __align__(16) bf16_t tmp[16];
#pragma unroll
    for (int i = 0; i < 16; ++i) {
      float v = qk[1][m * 64 + nb + i] * rnorm[1][nb + i];
      part += v;
      tmp[i] = (bf16_t)v;
    }
    bf16_t* dst = Kb + (size_t)b * CQ * NPOS + (size_t)m * NPOS + n0 + nb;
    *(v8bf*)(dst)     = *(const v8bf*)(tmp);
    *(v8bf*)(dst + 8) = *(const v8bf*)(tmp + 8);
    atomicAdd(&ksum[b * CQ + m], part);
  }
}

// ---------------------------------------------------------------------------
// T[b][m][c'] = sum_n Khat[m][n] * x[c'][n]   (GEMM, K-dim = n = 9216)
//   grid = B * 8 chunks * 16 groups, 8 waves/block, 1 tile/wave.
//   mtile==0 waves also accumulate xs[c'] = sum_n x[c'][n] from the f32
//   B values (before bf16 conversion) -> no separate colsum pass over x.
// ---------------------------------------------------------------------------
__global__ void tgemm_kernel(const float* __restrict__ x,
                             const bf16_t* __restrict__ Kb,
                             float* T, float* xs) {
  int bid = blockIdx.x;
  int b = bid / 128, rem = bid % 128;
  int chunk = rem / 16, tg = rem % 16;
  int wave = threadIdx.x >> 5, lane = threadIdx.x & 31;
  int tile = tg * 8 + wave;             // 0..127 = 4 mtiles x 32 c'tiles
  int mtile = tile & 3, ctile = tile >> 2;

  const bf16_t* A  = Kb + (size_t)b * CQ * NPOS + (size_t)(mtile * 16) * NPOS;
  const float*  Bp = x + (size_t)b * CCH * NPOS + (size_t)(ctile * 16) * NPOS;
  int nbeg = chunk * (NPOS / 8);        // 1152-wide K split

  int jl = lane & 15, kb2 = (lane >> 4) * 16;
  const float* Bq = Bp + (size_t)jl * NPOS + nbeg + kb2;

  v8f acc = {};
  v4f xsa = {0.0f, 0.0f, 0.0f, 0.0f};
#pragma unroll 2
  for (int k = 0; k < NPOS / 8; k += 32) {
    v16bf a = load_frag_a(A + nbeg + k, NPOS, lane);
    const float* q = Bq + k;
    v4f f0 = *(const v4f*)(q);
    v4f f1 = *(const v4f*)(q + 4);
    v4f f2 = *(const v4f*)(q + 8);
    v4f f3 = *(const v4f*)(q + 12);
    if (mtile == 0) xsa += f0 + f1 + f2 + f3;   // column-sum side product
    v16bf bf;
#pragma unroll
    for (int i = 0; i < 4; ++i) {
      bf[i]      = (bf16_t)f0[i];
      bf[4 + i]  = (bf16_t)f1[i];
      bf[8 + i]  = (bf16_t)f2[i];
      bf[12 + i] = (bf16_t)f3[i];
    }
    acc = wmma_bf16(a, bf, acc);
  }
  int mh = (lane >> 4) * 8;
  float* Td = T + (size_t)b * CQ * CCH;
#pragma unroll
  for (int r = 0; r < 8; ++r) {
    int m = mtile * 16 + mh + r;
    atomicAdd(&Td[m * CCH + ctile * 16 + jl], acc[r]);
  }
  if (mtile == 0)
    atomicAdd(&xs[b * CCH + ctile * 16 + jl],
              xsa[0] + xsa[1] + xsa[2] + xsa[3]);
}

// ---------------------------------------------------------------------------
// matrix[m][c] = sum_c' T[m][c'] wv[c][c'] + bv[c]*ksum[m]  -> mt[b][c][m] bf16
// vsum[c]     = sum_c' wv[c][c'] xs[c'] + N*bv[c]
// ---------------------------------------------------------------------------
__global__ void matrix_kernel(const float* __restrict__ T,
                              const float* __restrict__ wv,
                              const float* __restrict__ bv,
                              const float* __restrict__ ksum,
                              const float* __restrict__ xs,
                              bf16_t* mt, float* vsum) {
  int b = blockIdx.x / CCH, c = blockIdx.x % CCH;
  int t = threadIdx.x;
  __shared__ float wrow[CCH];
  __shared__ float xsl[CCH];
  __shared__ float red[256];
  __shared__ float mred[64][4];
  for (int i = t; i < CCH; i += 256) {
    wrow[i] = wv[(size_t)c * CCH + i];
    xsl[i]  = xs[b * CCH + i];
  }
  __syncthreads();
  {
    int m = t >> 2, q = t & 3;
    const float* Trow = T + ((size_t)b * CQ + m) * CCH;
    float s = 0.0f;
    for (int j = q * 128; j < q * 128 + 128; ++j) s += Trow[j] * wrow[j];
    mred[m][q] = s;
  }
  red[t] = wrow[t * 2] * xsl[t * 2] + wrow[t * 2 + 1] * xsl[t * 2 + 1];
  __syncthreads();
  if (t < 64) {
    float v = mred[t][0] + mred[t][1] + mred[t][2] + mred[t][3]
            + bv[c] * ksum[b * CQ + t];
    mt[((size_t)b * CCH + c) * CQ + t] = (bf16_t)v;
  }
  for (int o = 128; o > 0; o >>= 1) {
    __syncthreads();
    if (t < o) red[t] += red[t + o];
  }
  if (t == 0) vsum[b * CCH + c] = red[0] + (float)NPOS * bv[c];
}

// tailor[b][n] = 1 / (N + sum_m Qhat[m][n]*(ksum[m]+eps))
__global__ void tailor_kernel(const bf16_t* __restrict__ Qbt,
                              const float* __restrict__ ksum, float* tailor) {
  int b = blockIdx.x / (NPOS / 256);
  int n = (blockIdx.x % (NPOS / 256)) * 256 + threadIdx.x;
  __shared__ float ks[CQ];
  if (threadIdx.x < CQ) ks[threadIdx.x] = ksum[b * CQ + threadIdx.x] + EPSV;
  __syncthreads();
  const v8bf* qv = (const v8bf*)(Qbt + ((size_t)b * NPOS + n) * 64);
  float s = 0.0f;
#pragma unroll
  for (int j = 0; j < 8; ++j) {
    v8bf v = qv[j];
#pragma unroll
    for (int i = 0; i < 8; ++i) s += (float)v[i] * ks[j * 8 + i];
  }
  tailor[b * NPOS + n] = 1.0f / ((float)NPOS + s);
}

// ---------------------------------------------------------------------------
// out[c][n] = x[c][n] + gamma*tailor[n]*(vsum[c] + sum_m mt[c][m]*Qhat[m][n])
//   R-GEMM M=512(c), N=9216(n), K=64(m). Each wave: one c-tile x 4 n-subtiles,
//   A fragments + vsum loaded once, 8 WMMAs, fused residual epilogue.
// ---------------------------------------------------------------------------
__global__ void out_kernel(const float* __restrict__ x,
                           const bf16_t* __restrict__ mt,
                           const bf16_t* __restrict__ Qbt,
                           const float* __restrict__ vsum,
                           const float* __restrict__ tailor,
                           const float* __restrict__ gamma,
                           float* __restrict__ out) {
  int bid = blockIdx.x;
  int b = bid / 576, rem = bid % 576;
  int wave = threadIdx.x >> 5, lane = threadIdx.x & 31;
  int grp = rem * 8 + wave;             // 0..4607 = 32 ctiles x 144 ngroups
  int ct = grp & 31, ng = grp >> 5;
  int c0 = ct * 16, n0g = ng * 64;

  const bf16_t* A = mt + ((size_t)b * CCH + c0) * CQ;     // [c][m], m contig
  v16bf a0 = load_frag_a(A, CQ, lane);                    // k = 0..31
  v16bf a1 = load_frag_a(A + 32, CQ, lane);               // k = 32..63

  float g = gamma[0];
  int nl = lane & 15, mh = (lane >> 4) * 8;
  float vs[8];
#pragma unroll
  for (int r = 0; r < 8; ++r) vs[r] = vsum[b * CCH + c0 + mh + r];

  const float* xb = x + (size_t)b * CCH * NPOS;
  float* ob = out + (size_t)b * CCH * NPOS;

#pragma unroll
  for (int sub = 0; sub < 4; ++sub) {
    int n0 = n0g + sub * 16;
    const bf16_t* Bq = Qbt + ((size_t)b * NPOS + n0) * CQ;  // [n][m], m contig
    v8f acc = {};
    acc = wmma_bf16(a0, load_frag_b(Bq, CQ, lane), acc);
    acc = wmma_bf16(a1, load_frag_b(Bq + 32, CQ, lane), acc);
    int n = n0 + nl;
    float tl = tailor[b * NPOS + n] * g;
#pragma unroll
    for (int r = 0; r < 8; ++r) {
      size_t off = (size_t)(c0 + mh + r) * NPOS + n;
      ob[off] = xb[off] + tl * (vs[r] + acc[r]);
    }
  }
}

// ---------------------------------------------------------------------------
extern "C" void kernel_launch(void* const* d_in, const int* in_sizes, int n_in,
                              void* d_out, int out_size, void* d_ws, size_t ws_size,
                              hipStream_t stream) {
  (void)in_sizes; (void)n_in; (void)out_size; (void)ws_size;
  const float* x     = (const float*)d_in[0];
  const float* wq    = (const float*)d_in[1];
  const float* bq    = (const float*)d_in[2];
  const float* wk    = (const float*)d_in[3];
  const float* bk    = (const float*)d_in[4];
  const float* wv    = (const float*)d_in[5];
  const float* bv    = (const float*)d_in[6];
  const float* gamma = (const float*)d_in[7];
  float* out = (float*)d_out;

  // workspace layout (bytes); ksum/xs/T kept adjacent for one zero-init pass
  char* w = (char*)d_ws;
  size_t off = 0;
  bf16_t* Qbt  = (bf16_t*)(w + off); off += (size_t)BATCH * NPOS * CQ * 2;   // 9.4 MB
  bf16_t* Kb   = (bf16_t*)(w + off); off += (size_t)BATCH * CQ * NPOS * 2;   // 9.4 MB
  float*  ksum = (float*) (w + off); off += (size_t)BATCH * CQ * 4;
  float*  xs   = (float*) (w + off); off += (size_t)BATCH * CCH * 4;
  float*  T    = (float*) (w + off); off += (size_t)BATCH * CQ * CCH * 4;    // 1 MB
  bf16_t* mt   = (bf16_t*)(w + off); off += (size_t)BATCH * CCH * CQ * 2;
  float*  vsum = (float*) (w + off); off += (size_t)BATCH * CCH * 4;
  float*  tlr  = (float*) (w + off); off += (size_t)BATCH * NPOS * 4;
  bf16_t* wqb  = (bf16_t*)(w + off); off += (size_t)CQ * CCH * 2;
  bf16_t* wkb  = (bf16_t*)(w + off); off += (size_t)CQ * CCH * 2;           // ~20.9 MB

  int zcount = BATCH * CQ + BATCH * CCH + BATCH * CQ * CCH;  // ksum+xs+T
  zero_kernel<<<(zcount + 255) / 256, 256, 0, stream>>>(ksum, zcount);
  wcvt_kernel<<<(CQ * CCH + 255) / 256, 256, 0, stream>>>(wq, wk, wqb, wkb);
  qk_kernel<<<BATCH * (NPOS / 64), 256, 0, stream>>>(x, wqb, wkb, bq, bk, Qbt, Kb, ksum);
  tgemm_kernel<<<BATCH * 128, 256, 0, stream>>>(x, Kb, T, xs);
  matrix_kernel<<<BATCH * CCH, 256, 0, stream>>>(T, wv, bv, ksum, xs, mt, vsum);
  tailor_kernel<<<BATCH * (NPOS / 256), 256, 0, stream>>>(Qbt, ksum, tlr);
  out_kernel<<<BATCH * 576, 256, 0, stream>>>(x, mt, Qbt, vsum, tlr, gamma, out);
}
